// Mamba2_45732811768047
// MI455X (gfx1250) — compile-verified, matrix operand order checked by reading, not run on hardware
//
#include <hip/hip_runtime.h>
#include <hip/hip_bf16.h>
#include <math.h>

typedef __bf16 bf16_t;
typedef __attribute__((ext_vector_type(16))) __bf16 v16bf;
typedef __attribute__((ext_vector_type(8)))  float  v8f;

#define D_MODEL   2048
#define D_SSM     4096
#define D_STATE   128
#define NHEADS    64
#define HEADDIM   64
#define CHUNK     64
#define D_IN_PROJ 8512
#define CONV_DIM  4352
#define BATCH     2
#define SEQLEN    4096
#define NCHUNK    (SEQLEN / CHUNK)     // 64
#define ROWS      (BATCH * SEQLEN)     // 8192

#if defined(__has_builtin)
#if __has_builtin(__builtin_amdgcn_tensor_load_to_lds) && \
    __has_builtin(__builtin_amdgcn_s_wait_tensorcnt)
#define USE_TDM 1
#endif
#endif

// ---------------- WMMA helpers (CDNA5: wave32, 16x16x32 bf16 -> f32) ----------------

__device__ __forceinline__ v8f vzero8() {
  v8f z;
#pragma unroll
  for (int i = 0; i < 8; ++i) z[i] = 0.f;
  return z;
}

__device__ __forceinline__ v8f wmma_bf16(v16bf a, v16bf b, v8f c) {
  // (neg_a, A, neg_b, B, c_mod, C, reuse_a, reuse_b)
  return __builtin_amdgcn_wmma_f32_16x16x32_bf16(false, a, false, b, (short)0, c,
                                                 false, false);
}

// Fragment load, operand stored as [16 rows][K contiguous] (A-matrix, or B stored N-major)
__device__ __forceinline__ v16bf load_frag_rowK(const bf16_t* base, int ldk, int lane) {
  const bf16_t* p = base + (size_t)(lane & 15) * ldk + ((lane >> 4) << 4);
  return *(const v16bf*)p;
}

// Fragment load, operand stored as [K rows][N contiguous] (gathered, strided)
__device__ __forceinline__ v16bf load_frag_colN(const bf16_t* base, int ldn, int lane) {
  const bf16_t* p = base + (size_t)((lane >> 4) << 4) * ldn + (lane & 15);
  v16bf r;
#pragma unroll
  for (int i = 0; i < 16; ++i) r[i] = p[(size_t)i * ldn];
  return r;
}

#ifdef USE_TDM
// ---------------- Tensor Data Mover: 2-D bf16 tile -> LDS ----------------
typedef __attribute__((ext_vector_type(4))) unsigned int u32x4;
typedef __attribute__((ext_vector_type(8))) int i32x8;
typedef __attribute__((ext_vector_type(4))) int i32x4;

// D# per CDNA5 ISA: group0 = {count/type, lds_addr, global_addr},
// group1 = {data_size, tensor dims, tile dims, dim0 stride}; groups 2/3 zero (2-D).
// This toolchain exposes the 6-arg builtin (extra trailing int32x8 group; zeros).
__device__ __forceinline__ void tdm_load_tile_2d(const bf16_t* gsrc, void* lds_dst,
                                                 unsigned tile_k, unsigned tile_rows,
                                                 unsigned row_stride_elems) {
  unsigned long long ga = (unsigned long long)(size_t)gsrc;
  unsigned la = (unsigned)(size_t)lds_dst;   // generic LDS ptr: low 32 bits = LDS offset
  u32x4 g0;
  g0.x = 1u;                                            // count=1, user descriptor
  g0.y = la;                                            // lds_addr [63:32]
  g0.z = (unsigned)ga;                                  // global_addr [95:64]
  g0.w = (unsigned)((ga >> 32) & 0x01ffffffu) | (2u << 30);  // addr[56:32] | type=2
  i32x8 g1;
  g1[0] = (int)(1u << 16);                              // data_size=1 -> 2 bytes/elem
  g1[1] = (int)((tile_k & 0xffffu) << 16);              // tensor_dim0 lo16
  g1[2] = (int)(((tile_k >> 16) & 0xffffu) |
                ((tile_rows & 0xffffu) << 16));         // dim0 hi | tensor_dim1 lo
  g1[3] = (int)(((tile_rows >> 16) & 0xffffu) |
                ((tile_k & 0xffffu) << 16));            // dim1 hi | tile_dim0
  g1[4] = (int)(tile_rows & 0xffffu);                   // tile_dim1 (tile_dim2=0)
  g1[5] = (int)row_stride_elems;                        // tensor_dim0_stride lo32
  g1[6] = 0;                                            // stride hi | dim1_stride lo
  g1[7] = 0;
  i32x4 z4;
  z4[0] = 0; z4[1] = 0; z4[2] = 0; z4[3] = 0;
  i32x8 z8;
#pragma unroll
  for (int i = 0; i < 8; ++i) z8[i] = 0;
  __builtin_amdgcn_tensor_load_to_lds(g0, g1, z4, z4, z8, 0);
}
#endif

// ---------------- K1: fp32 -> bf16 conversion ----------------

__global__ void f32_to_bf16_kernel(const float* __restrict__ in, bf16_t* __restrict__ out,
                                   size_t n) {
  size_t i = (size_t)blockIdx.x * blockDim.x + threadIdx.x;
  size_t stride = (size_t)gridDim.x * blockDim.x;
  for (; i < n; i += stride) out[i] = (bf16_t)in[i];
}

// ---------------- K2/K8: GEMM  C[M,N] = A[M,K] * B[N,K]^T  (bf16 in, f32 out) ------
// Block: 256 thr = 8 waves (4x2), tile 128x64, per-wave 32x32 (2x2 WMMA accum)
// TDM variant: double-buffered LDS, wave 0 drives the Tensor Data Mover.

__global__ __launch_bounds__(256) void gemm_bf16nt(const bf16_t* __restrict__ A,
                                                   const bf16_t* __restrict__ B,
                                                   float* __restrict__ C,
                                                   int M, int N, int K) {
#ifdef USE_TDM
  __shared__ alignas(128) bf16_t As[2][128][32];
  __shared__ alignas(128) bf16_t Bs[2][64][32];
#else
  __shared__ alignas(128) bf16_t As[1][128][32];
  __shared__ alignas(128) bf16_t Bs[1][64][32];
#endif
  const int tid  = threadIdx.x;
  const int lane = tid & 31;
  const int wave = tid >> 5;
  const int wm = wave & 3;          // 0..3 -> M sub-tile
  const int wn = wave >> 2;         // 0..1 -> N sub-tile
  const int m0 = blockIdx.x * 128;
  const int n0 = blockIdx.y * 64;

  v8f acc[2][2];
#pragma unroll
  for (int i = 0; i < 2; ++i)
#pragma unroll
    for (int j = 0; j < 2; ++j) acc[i][j] = vzero8();

#ifdef USE_TDM
  const int nkt = K >> 5;
  if (wave == 0) {  // prime the pipeline: tile 0 into buffer 0
    tdm_load_tile_2d(A + (size_t)m0 * K, &As[0][0][0], 32u, 128u, (unsigned)K);
    tdm_load_tile_2d(B + (size_t)n0 * K, &Bs[0][0][0], 32u, 64u, (unsigned)K);
  }
  for (int kt = 0; kt < nkt; ++kt) {
    const int cur = kt & 1;
    if (wave == 0) __builtin_amdgcn_s_wait_tensorcnt((short)0);
    __syncthreads();  // publish: buffer `cur` ready, buffer `cur^1` reads retired
    if (wave == 0 && kt + 1 < nkt) {  // kick next tile while computing this one
      const int k0n = (kt + 1) << 5;
      tdm_load_tile_2d(A + (size_t)m0 * K + k0n, &As[cur ^ 1][0][0], 32u, 128u, (unsigned)K);
      tdm_load_tile_2d(B + (size_t)n0 * K + k0n, &Bs[cur ^ 1][0][0], 32u, 64u, (unsigned)K);
    }
    v16bf bfrag[2];
#pragma unroll
    for (int nt = 0; nt < 2; ++nt)
      bfrag[nt] = load_frag_rowK(&Bs[cur][wn * 32 + nt * 16][0], 32, lane);
#pragma unroll
    for (int mt = 0; mt < 2; ++mt) {
      v16bf afrag = load_frag_rowK(&As[cur][wm * 32 + mt * 16][0], 32, lane);
#pragma unroll
      for (int nt = 0; nt < 2; ++nt)
        acc[mt][nt] = wmma_bf16(afrag, bfrag[nt], acc[mt][nt]);
    }
  }
#else
  const int arow = tid >> 1, acol = (tid & 1) << 4;  // 128 rows x 32 (16 elem/thr)
  const int brow = tid >> 2, bcol = (tid & 3) << 3;  // 64 rows x 32 (8 elem/thr)
  for (int k0 = 0; k0 < K; k0 += 32) {
    __syncthreads();
    const uint4* ap = (const uint4*)(A + (size_t)(m0 + arow) * K + k0 + acol);
    *(uint4*)&As[0][arow][acol]     = ap[0];
    *(uint4*)&As[0][arow][acol + 8] = ap[1];
    *(uint4*)&Bs[0][brow][bcol] = *(const uint4*)(B + (size_t)(n0 + brow) * K + k0 + bcol);
    if (k0 + 32 < K) {
      __builtin_prefetch(A + (size_t)(m0 + arow) * K + k0 + 32 + acol, 0, 1);
      __builtin_prefetch(B + (size_t)(n0 + brow) * K + k0 + 32 + bcol, 0, 1);
    }
    __syncthreads();
    v16bf bfrag[2];
#pragma unroll
    for (int nt = 0; nt < 2; ++nt)
      bfrag[nt] = load_frag_rowK(&Bs[0][wn * 32 + nt * 16][0], 32, lane);
#pragma unroll
    for (int mt = 0; mt < 2; ++mt) {
      v16bf afrag = load_frag_rowK(&As[0][wm * 32 + mt * 16][0], 32, lane);
#pragma unroll
      for (int nt = 0; nt < 2; ++nt)
        acc[mt][nt] = wmma_bf16(afrag, bfrag[nt], acc[mt][nt]);
    }
  }
#endif

  const int hi = (lane >> 4) << 3;  // lanes 16-31 hold M = r+8
  const int nc = lane & 15;
#pragma unroll
  for (int mt = 0; mt < 2; ++mt)
#pragma unroll
    for (int nt = 0; nt < 2; ++nt)
#pragma unroll
      for (int r = 0; r < 8; ++r) {
        int m = m0 + wm * 32 + mt * 16 + r + hi;
        int n = n0 + wn * 32 + nt * 16 + nc;
        C[(size_t)m * N + n] = acc[mt][nt][r];
      }
}

// ---------------- K3a: depthwise causal conv(4) + SiLU ----------------

__global__ void conv_silu_kernel(const float* __restrict__ zx,
                                 const float* __restrict__ conv_w,
                                 const float* __restrict__ conv_b,
                                 float* __restrict__ xbc) {
  size_t idx = (size_t)blockIdx.x * blockDim.x + threadIdx.x;
  size_t total = (size_t)ROWS * CONV_DIM;
  if (idx >= total) return;
  int ch = (int)(idx % CONV_DIM);
  size_t bl = idx / CONV_DIM;
  int l = (int)(bl % SEQLEN);
  size_t brow = bl - l;
  float acc = conv_b[ch];
#pragma unroll
  for (int k = 0; k < 4; ++k) {
    int ll = l - 3 + k;
    if (ll >= 0)
      acc += zx[(brow + ll) * (size_t)D_IN_PROJ + D_SSM + ch] * conv_w[ch * 4 + k];
  }
  xbc[idx] = acc / (1.f + expf(-acc));   // SiLU
}

// ---------------- K3b: dt softplus, within-chunk cumsum of dt*A, chunk decay -------

__global__ void dt_cumsum_kernel(const float* __restrict__ zx,
                                 const float* __restrict__ dt_bias,
                                 const float* __restrict__ A_log,
                                 float* __restrict__ dtsp,
                                 float* __restrict__ acum,
                                 float* __restrict__ cdec) {
  int t = blockIdx.x * blockDim.x + threadIdx.x;  // (b, c, h)
  if (t >= BATCH * NCHUNK * NHEADS) return;
  int h = t % NHEADS;
  int c = (t / NHEADS) % NCHUNK;
  int b = t / (NHEADS * NCHUNK);
  const float A = -expf(A_log[h]);
  const float bias = dt_bias[h];
  float run = 0.f;
  for (int l = 0; l < CHUNK; ++l) {
    size_t row = (size_t)b * SEQLEN + (size_t)c * CHUNK + l;
    float x = zx[row * D_IN_PROJ + (D_IN_PROJ - NHEADS) + h] + bias;
    float dtv = (x > 20.f) ? x : log1pf(expf(x));  // softplus
    run += dtv * A;
    dtsp[row * NHEADS + h] = dtv;
    acum[row * NHEADS + h] = run;
  }
  cdec[t] = expf(run);
}

// ---------------- K4: per-(b,chunk,head) intra-chunk WMMA ----------------
// CB = Cc·Bc^T ; Y_diag = (CB*L*dt)·x + D*x ; states = (decay*dt*x)^T·Bc

__global__ __launch_bounds__(128) void chunk_kernel(
    const float* __restrict__ xbc,    // [ROWS][CONV_DIM]
    const float* __restrict__ dtsp,   // [ROWS][NHEADS]
    const float* __restrict__ acum,   // [ROWS][NHEADS]
    const float* __restrict__ Dp,     // [NHEADS]
    float* __restrict__ y,            // [ROWS][NHEADS][HEADDIM]
    float* __restrict__ states) {     // [B][NC][H][HEADDIM][D_STATE]
  __shared__ alignas(64) bf16_t sCc[64][128];  // later aliased: sM (rows 0-31), sX (rows 32-63)
  __shared__ alignas(64) bf16_t sBc[64][128];
  __shared__ alignas(64) float  sCB[64][64];
  __shared__ alignas(64) bf16_t sXwT[64][64];
  __shared__ float sAc[64], sDt[64], sW[64];

  const int blk = blockIdx.x;
  const int h = blk % NHEADS;
  const int c = (blk / NHEADS) % NCHUNK;
  const int b = blk / (NHEADS * NCHUNK);
  const size_t row0 = (size_t)b * SEQLEN + (size_t)c * CHUNK;
  const int tid = threadIdx.x;
  const int lane = tid & 31;
  const int wave = tid >> 5;  // 0..3

  if (tid < 64) {
    sAc[tid] = acum[(row0 + tid) * NHEADS + h];
    sDt[tid] = dtsp[(row0 + tid) * NHEADS + h];
  }
  for (int i = 0; i < 64; ++i) {          // stage Bc, Cc as bf16
    int flat = tid * 64 + i;
    int l = flat >> 7, n = flat & 127;
    const float* src = xbc + (row0 + l) * CONV_DIM;
    sBc[l][n] = (bf16_t)src[D_SSM + n];
    sCc[l][n] = (bf16_t)src[D_SSM + D_STATE + n];
  }
  __syncthreads();
  if (tid < 64) sW[tid] = expf(sAc[63] - sAc[tid]) * sDt[tid];

  // CB = Cc (64x128) @ Bc^T -> 64x64 fp32
  {
    const int mt = wave;
    const int hi = (lane >> 4) << 3, nc = lane & 15;
    for (int nt = 0; nt < 4; ++nt) {
      v8f acc = vzero8();
#pragma unroll
      for (int ks = 0; ks < 4; ++ks) {
        v16bf a  = load_frag_rowK(&sCc[mt * 16][ks * 32], 128, lane);
        v16bf bb = load_frag_rowK(&sBc[nt * 16][ks * 32], 128, lane);
        acc = wmma_bf16(a, bb, acc);
      }
#pragma unroll
      for (int r = 0; r < 8; ++r) sCB[mt * 16 + r + hi][nt * 16 + nc] = acc[r];
    }
  }
  __syncthreads();

  // Build masked/decayed M, x tile, and decayed x^T (reuse sCc storage)
  bf16_t(*sM)[64] = (bf16_t(*)[64]) & sCc[0][0];
  bf16_t(*sX)[64] = (bf16_t(*)[64]) & sCc[32][0];
  for (int i = 0; i < 32; ++i) {
    int flat = tid * 32 + i;
    int l = flat >> 6, s = flat & 63;
    float mv = (s <= l) ? sCB[l][s] * expf(sAc[l] - sAc[s]) * sDt[s] : 0.f;
    sM[l][s] = (bf16_t)mv;
    float xv = xbc[(row0 + l) * CONV_DIM + h * HEADDIM + s];
    sX[l][s] = (bf16_t)xv;
    sXwT[s][l] = (bf16_t)(xv * sW[l]);
  }
  __syncthreads();

  // Y_diag = M (64x64) @ x (64x64)  (+ D*x), write y
  {
    const float Dh = Dp[h];
    const int mt = wave;
    const int hi = (lane >> 4) << 3, nc = lane & 15;
    for (int nt = 0; nt < 4; ++nt) {
      v8f acc = vzero8();
#pragma unroll
      for (int ks = 0; ks < 2; ++ks) {
        v16bf a  = load_frag_rowK(&sM[mt * 16][ks * 32], 64, lane);
        v16bf bb = load_frag_colN(&sX[ks * 32][nt * 16], 64, lane);
        acc = wmma_bf16(a, bb, acc);
      }
#pragma unroll
      for (int r = 0; r < 8; ++r) {
        int l = mt * 16 + r + hi;
        int p = nt * 16 + nc;
        y[((row0 + l) * NHEADS + h) * HEADDIM + p] = acc[r] + Dh * (float)sX[l][p];
      }
    }
  }

  // states = xW^T (64x64) @ Bc (64x128) -> HEADDIM x D_STATE
  {
    const int pt = wave;
    const int hi = (lane >> 4) << 3, nc = lane & 15;
    const size_t sbase = (((size_t)b * NCHUNK + c) * NHEADS + h) * (size_t)(HEADDIM * D_STATE);
    for (int nt = 0; nt < 8; ++nt) {
      v8f acc = vzero8();
#pragma unroll
      for (int ks = 0; ks < 2; ++ks) {
        v16bf a  = load_frag_rowK(&sXwT[pt * 16][ks * 32], 64, lane);
        v16bf bb = load_frag_colN(&sBc[ks * 32][nt * 16], 128, lane);
        acc = wmma_bf16(a, bb, acc);
      }
#pragma unroll
      for (int r = 0; r < 8; ++r) {
        int p = pt * 16 + r + hi;
        int n = nt * 16 + nc;
        states[sbase + (size_t)p * D_STATE + n] = acc[r];
      }
    }
  }
}

// ---------------- K5: inter-chunk scan, per (b,h); prev written in place ----------

__global__ __launch_bounds__(256) void scan_kernel(float* __restrict__ states,
                                                   const float* __restrict__ cdec) {
  const int b = blockIdx.x >> 6;
  const int h = blockIdx.x & 63;
  const int t = threadIdx.x;
  float4 carry[8];
#pragma unroll
  for (int i = 0; i < 8; ++i) carry[i] = make_float4(0.f, 0.f, 0.f, 0.f);
  for (int c = 0; c < NCHUNK; ++c) {
    const float cd = cdec[((size_t)b * NCHUNK + c) * NHEADS + h];
    float4* p = (float4*)(states +
                          (((size_t)b * NCHUNK + c) * NHEADS + h) * (size_t)(HEADDIM * D_STATE) +
                          (size_t)t * 32);
#pragma unroll
    for (int i = 0; i < 8; ++i) {
      float4 s = p[i];
      float4 pr = carry[i];
      p[i] = pr;  // prev[c] = carry before update
      carry[i].x = pr.x * cd + s.x;
      carry[i].y = pr.y * cd + s.y;
      carry[i].z = pr.z * cd + s.z;
      carry[i].w = pr.w * cd + s.w;
    }
  }
}

// ---------------- K6: Y_off = (e^Acum * Cc) @ prev^T, accumulate into y -----------

__global__ __launch_bounds__(128) void yoff_kernel(const float* __restrict__ xbc,
                                                   const float* __restrict__ acum,
                                                   const float* __restrict__ states,  // prev
                                                   float* __restrict__ y) {
  __shared__ alignas(64) bf16_t sCe[64][128];
  __shared__ alignas(64) bf16_t sPr[64][128];  // [p][n]
  __shared__ float sEA[64];

  const int blk = blockIdx.x;
  const int h = blk % NHEADS;
  const int c = (blk / NHEADS) % NCHUNK;
  const int b = blk / (NHEADS * NCHUNK);
  const size_t row0 = (size_t)b * SEQLEN + (size_t)c * CHUNK;
  const int tid = threadIdx.x;
  const int lane = tid & 31;
  const int wave = tid >> 5;

  if (tid < 64) sEA[tid] = expf(acum[(row0 + tid) * NHEADS + h]);
  __syncthreads();
  const size_t sbase = (((size_t)b * NCHUNK + c) * NHEADS + h) * (size_t)(HEADDIM * D_STATE);
  for (int i = 0; i < 64; ++i) {
    int flat = tid * 64 + i;
    int l = flat >> 7, n = flat & 127;
    sCe[l][n] = (bf16_t)(xbc[(row0 + l) * CONV_DIM + D_SSM + D_STATE + n] * sEA[l]);
    sPr[l][n] = (bf16_t)states[sbase + (size_t)l * D_STATE + n];  // l plays role of p
  }
  __syncthreads();

  const int mt = wave;
  const int hi = (lane >> 4) << 3, nc = lane & 15;
  for (int nt = 0; nt < 4; ++nt) {
    v8f acc = vzero8();
#pragma unroll
    for (int ks = 0; ks < 4; ++ks) {
      v16bf a  = load_frag_rowK(&sCe[mt * 16][ks * 32], 128, lane);
      v16bf bb = load_frag_rowK(&sPr[nt * 16][ks * 32], 128, lane);  // [N=p][K=n]
      acc = wmma_bf16(a, bb, acc);
    }
#pragma unroll
    for (int r = 0; r < 8; ++r) {
      int l = mt * 16 + r + hi;
      int p = nt * 16 + nc;
      size_t off = ((row0 + l) * NHEADS + h) * HEADDIM + p;
      y[off] += acc[r];
    }
  }
}

// ---------------- K7: gating (SiLU(z)) + RMSNorm -> bf16 ----------------

__global__ __launch_bounds__(256) void gate_norm_kernel(const float* __restrict__ y,
                                                        const float* __restrict__ zx,
                                                        const float* __restrict__ nw,
                                                        bf16_t* __restrict__ yb) {
  const int row = blockIdx.x;
  const int t = threadIdx.x;
  __shared__ float red[256];
  float vals[16];
  float ss = 0.f;
#pragma unroll
  for (int i = 0; i < 16; ++i) {
    int ci = t + i * 256;
    float z = zx[(size_t)row * D_IN_PROJ + ci];
    float yg = y[(size_t)row * D_SSM + ci] * (z / (1.f + expf(-z)));
    vals[i] = yg;
    ss += yg * yg;
  }
  red[t] = ss;
  __syncthreads();
  for (int s = 128; s > 0; s >>= 1) {
    if (t < s) red[t] += red[t + s];
    __syncthreads();
  }
  const float scale = rsqrtf(red[0] / (float)D_SSM + 1e-5f);
#pragma unroll
  for (int i = 0; i < 16; ++i) {
    int ci = t + i * 256;
    yb[(size_t)row * D_SSM + ci] = (bf16_t)(vals[i] * scale * nw[ci]);
  }
}

// ---------------- host side ----------------

extern "C" void kernel_launch(void* const* d_in, const int* in_sizes, int n_in,
                              void* d_out, int out_size, void* d_ws, size_t ws_size,
                              hipStream_t stream) {
  (void)in_sizes; (void)n_in; (void)out_size; (void)ws_size;
  const float* u      = (const float*)d_in[0];
  const float* W_in   = (const float*)d_in[1];
  const float* conv_w = (const float*)d_in[2];
  const float* conv_b = (const float*)d_in[3];
  const float* dtb    = (const float*)d_in[4];
  const float* A_log  = (const float*)d_in[5];
  const float* Dp     = (const float*)d_in[6];
  const float* nw     = (const float*)d_in[7];
  const float* W_out  = (const float*)d_in[8];

  size_t off = 0;
  char* base = (char*)d_ws;
  auto alloc = [&](size_t bytes) -> void* {
    void* p = base + off;
    off += (bytes + 255) & ~(size_t)255;
    return p;
  };
  bf16_t* u_bf    = (bf16_t*)alloc((size_t)ROWS * D_MODEL * 2);
  bf16_t* win_bf  = (bf16_t*)alloc((size_t)D_IN_PROJ * D_MODEL * 2);
  bf16_t* wout_bf = (bf16_t*)alloc((size_t)D_MODEL * D_SSM * 2);
  float*  zx      = (float*)alloc((size_t)ROWS * D_IN_PROJ * 4);
  float*  xbc     = (float*)alloc((size_t)ROWS * CONV_DIM * 4);
  float*  dtsp    = (float*)alloc((size_t)ROWS * NHEADS * 4);
  float*  acum    = (float*)alloc((size_t)ROWS * NHEADS * 4);
  float*  cdec    = (float*)alloc((size_t)BATCH * NCHUNK * NHEADS * 4);
  float*  st      = (float*)alloc((size_t)BATCH * NCHUNK * NHEADS * HEADDIM * D_STATE * 4);
  float*  ybuf    = (float*)alloc((size_t)ROWS * D_SSM * 4);
  bf16_t* ynorm   = (bf16_t*)alloc((size_t)ROWS * D_SSM * 2);

  // 1) fp32 -> bf16 conversions
  f32_to_bf16_kernel<<<4096, 256, 0, stream>>>(u, u_bf, (size_t)ROWS * D_MODEL);
  f32_to_bf16_kernel<<<4096, 256, 0, stream>>>(W_in, win_bf, (size_t)D_IN_PROJ * D_MODEL);
  f32_to_bf16_kernel<<<4096, 256, 0, stream>>>(W_out, wout_bf, (size_t)D_MODEL * D_SSM);

  // 2) in-projection GEMM: zx[8192, 8512]
  gemm_bf16nt<<<dim3(ROWS / 128, D_IN_PROJ / 64), 256, 0, stream>>>(
      u_bf, win_bf, zx, ROWS, D_IN_PROJ, D_MODEL);

  // 3) conv + SiLU ; dt softplus + chunk cumsum + chunk decay
  {
    size_t total = (size_t)ROWS * CONV_DIM;
    conv_silu_kernel<<<(unsigned)((total + 255) / 256), 256, 0, stream>>>(zx, conv_w, conv_b, xbc);
  }
  dt_cumsum_kernel<<<(BATCH * NCHUNK * NHEADS + 255) / 256, 256, 0, stream>>>(
      zx, dtb, A_log, dtsp, acum, cdec);

  // 4) intra-chunk WMMA: Y_diag + D*x and chunk states
  chunk_kernel<<<BATCH * NCHUNK * NHEADS, 128, 0, stream>>>(xbc, dtsp, acum, Dp, ybuf, st);

  // 5) inter-chunk scan (prev states written in place)
  scan_kernel<<<BATCH * NHEADS, 256, 0, stream>>>(st, cdec);

  // 6) Y_off accumulate
  yoff_kernel<<<BATCH * NCHUNK * NHEADS, 128, 0, stream>>>(xbc, acum, st, ybuf);

  // 7) gate + RMSNorm -> bf16
  gate_norm_kernel<<<ROWS, 256, 0, stream>>>(ybuf, zx, nw, ynorm);

  // 8) out-projection GEMM -> d_out (f32)
  gemm_bf16nt<<<dim3(ROWS / 128, D_MODEL / 64), 256, 0, stream>>>(
      ynorm, wout_bf, (float*)d_out, ROWS, D_MODEL, D_SSM);
}
